// Encoder_28630251995817
// MI455X (gfx1250) — compile-verified, matrix operand order checked by reading, not run on hardware
//
#include <hip/hip_runtime.h>
#include <hip/hip_fp16.h>
#include <math.h>

typedef __attribute__((ext_vector_type(16))) _Float16 v16h;
typedef __attribute__((ext_vector_type(8)))  float    v8f;

#define OMEGA_ENC 0.01f
#define OMEGA_K   0.1f
#define H 64
#define F_IN 8
#define LAT 32
#define FDIM 67   // H + 3

// ---------------------------------------------------------------- utilities
__global__ void k_zero(float* __restrict__ p, int n) {
    int i = blockIdx.x * blockDim.x + threadIdx.x;
    if (i < n) p[i] = 0.0f;
}

// ---------------------------------------------------------------- encoder: h = sin(0.01*(x@W+b))
__global__ void k_encode(const float* __restrict__ x, const float* __restrict__ w,
                         const float* __restrict__ b, float* __restrict__ h, int N) {
    int id = blockIdx.x * blockDim.x + threadIdx.x;
    if (id >= N * H) return;
    int n = id >> 6, c = id & 63;
    float s = b[c];
#pragma unroll
    for (int f = 0; f < F_IN; ++f) s += x[n * F_IN + f] * w[f * H + c];
    h[id] = __sinf(OMEGA_ENC * s);
}

// ---------------------------------------------------------------- per-channel angle tables (f16)
// tabc[c][h] = cos(omega_k*c*w0[3][h]), tabs_[c][h] = sin(...)
__global__ void k_tabs(const float* __restrict__ w0,
                       __half* __restrict__ tabc, __half* __restrict__ tabs_) {
    int id = blockIdx.x * blockDim.x + threadIdx.x;   // 64*64
    if (id >= H * H) return;
    int c = id >> 6, hh = id & 63;
    float ang = OMEGA_K * (float)c * w0[3 * H + hh];
    tabc[id] = __float2half(__cosf(ang));
    tabs_[id] = __float2half(__sinf(ang));
}

// ---------------------------------------------------------------- edge prep
// per edge: geometry -> base -> f16 {sin,cos}(omega*base); u = x_j @ w2^T; xbsum scatter
__global__ __launch_bounds__(64)
void k_edge_prep(const int* __restrict__ ei, int E, int Etot,
                 const float* __restrict__ pos, const float* __restrict__ xfH,
                 const float* __restrict__ w0, const float* __restrict__ b0,
                 const float* __restrict__ w2, const float* __restrict__ b2,
                 __half* __restrict__ sb_h, __half* __restrict__ cb_h,
                 float* __restrict__ u,
                 int* __restrict__ dstb, int* __restrict__ srcb,
                 float* __restrict__ xbsum) {
    int e = blockIdx.x;
    if (e >= Etot) return;
    int hh = threadIdx.x;
    int src, dst;
    if (e < E) { src = ei[e]; dst = ei[E + e]; }
    else       { src = dst = e - E; }           // appended self loops
    float r0 = pos[dst * 3 + 0] - pos[src * 3 + 0];
    float r1 = pos[dst * 3 + 1] - pos[src * 3 + 1];
    float r2 = pos[dst * 3 + 2] - pos[src * 3 + 2];
    float rho = 0.f, th = 0.f, ph = 0.f;
    if (!(r0 == 0.f && r1 == 0.f && r2 == 0.f)) {
        rho = sqrtf(r0 * r0 + r1 * r1 + r2 * r2);
        th  = atan2f(r1, r0);
        float zc = fminf(1.f, fmaxf(-1.f, r2 / rho));
        ph  = asinf(zc);
    }
    const float INVPI = 0.31830988618379067f;
    float base = rho * w0[hh] + (th * INVPI) * w0[H + hh] + (ph * INVPI) * w0[2 * H + hh] + b0[hh];
    float ang = OMEGA_K * base;
    sb_h[(size_t)e * H + hh] = __float2half(__sinf(ang));
    cb_h[(size_t)e * H + hh] = __float2half(__cosf(ang));
    const float* xr = xfH + (size_t)src * H;
    float s = w2[hh * FDIM + 64] * pos[src * 3 + 0]
            + w2[hh * FDIM + 65] * pos[src * 3 + 1]
            + w2[hh * FDIM + 66] * pos[src * 3 + 2];
#pragma unroll 8
    for (int f = 0; f < H; ++f) s += w2[hh * FDIM + f] * xr[f];
    u[(size_t)e * H + hh] = s;
    if (hh == 0) {
        float xb = b2[64] * pos[src * 3 + 0] + b2[65] * pos[src * 3 + 1] + b2[66] * pos[src * 3 + 2];
#pragma unroll 8
        for (int f = 0; f < H; ++f) xb += b2[f] * xr[f];
        atomicAdd(xbsum + dst, xb);
        dstb[e] = dst;
        srcb[e] = src;
    }
}

// ---------------------------------------------------------------- WMMA conv core
// block = 256 (8 waves); block handles one 32-edge tile (two 16-row M-tiles);
// wave w handles channels w*8..w*8+7.
__global__ __launch_bounds__(256)
void k_conv_wmma(const __half* __restrict__ sb_h,   // [Etot][64]
                 const __half* __restrict__ cb_h,   // [Etot][64]
                 const float*  __restrict__ u,      // [Etot][64]
                 const int*    __restrict__ dstbuf, // [Etot]
                 const __half* __restrict__ tabc,   // [64][64]
                 const __half* __restrict__ tabs_,  // [64][64]
                 const float*  __restrict__ w1,     // [64][64]
                 const float*  __restrict__ b1,     // [64]
                 float* __restrict__ outacc,        // [N][64] (atomic accum)
                 int Etot) {
    __shared__ __half s_tc[H * H];        // 8 KB
    __shared__ __half s_ts[H * H];        // 8 KB
    __shared__ __half s_sb[32][68];       // padded rows (stride 34 dwords) -> conflict-free
    __shared__ __half s_cb[32][68];
    __shared__ float  s_u[32][68];
    __shared__ int    s_dst[32];

    const int tid  = threadIdx.x;
    const int lane = tid & 31;
    const int wave = tid >> 5;
    const int hi   = lane >> 4;
    const int ln   = lane & 15;
    const int tileBase = blockIdx.x * 32;

    // stage tables (dword copies)
    {
        const unsigned* gc = (const unsigned*)tabc;
        const unsigned* gs = (const unsigned*)tabs_;
        unsigned* lc = (unsigned*)s_tc;
        unsigned* ls = (unsigned*)s_ts;
        for (int i = tid; i < H * H / 2; i += 256) { lc[i] = gc[i]; ls[i] = gs[i]; }
    }
    // stage edge tile
    for (int r = wave; r < 32; r += 8) {
        int e = tileBase + r;
        bool ok = (e < Etot);
        const __half2* sr = (const __half2*)(sb_h + (size_t)e * H);
        const __half2* cr = (const __half2*)(cb_h + (size_t)e * H);
        __half2 z = __float2half2_rn(0.f);
        ((__half2*)&s_sb[r][0])[lane] = ok ? sr[lane] : z;
        ((__half2*)&s_cb[r][0])[lane] = ok ? cr[lane] : z;
        const float* ur = u + (size_t)e * H;
#pragma unroll
        for (int q2 = lane; q2 < H; q2 += 32) s_u[r][q2] = ok ? ur[q2] : 0.f;
        if (lane == 0) s_dst[r] = ok ? dstbuf[e] : -1;
    }

    // B fragments (w1 -> f16), resident in registers. 32x16 B layout:
    // lane: N = ln + 16*t, half (j,p): Klocal = (j<4?0:16) + (hi?8:0) + (j&3)*2 + p
    v16h bf[4][2];
    float b1v[4];
#pragma unroll
    for (int t = 0; t < 4; ++t) {
        int N = t * 16 + ln;
        b1v[t] = b1[N];
#pragma unroll
        for (int s = 0; s < 2; ++s)
#pragma unroll
            for (int j = 0; j < 8; ++j)
#pragma unroll
                for (int p = 0; p < 2; ++p) {
                    int Kl = ((j & 4) ? 16 : 0) + (hi ? 8 : 0) + ((j & 3) << 1) + p;
                    int K  = s * 32 + Kl;
                    bf[t][s][j * 2 + p] = (_Float16)w1[K * H + N];
                }
    }
    if (lane == 0 && tileBase + 32 < Etot)
        __builtin_prefetch(sb_h + (size_t)(tileBase + 32) * H, 0, 1);

    __syncthreads();

#pragma unroll 1
    for (int q = 0; q < 8; ++q) {
        int c = wave * 8 + q;
        const __half* tcr = s_tc + c * H;
        const __half* tsr = s_ts + c * H;
#pragma unroll 1
        for (int mt = 0; mt < 2; ++mt) {
            const int rowb = mt * 16;
            const int m = rowb + ln;   // A row (16-bit A 16x32: both lane halves M=0..15)
            // Build f16 A fragments: h0 = sinB*cosT + cosB*sinT, two K's per pk op.
            union { v16h v; __half2 p[8]; } A0, A1;
#pragma unroll
            for (int j = 0; j < 8; ++j) {
                int Kl = ((j & 4) ? 16 : 0) + (hi ? 8 : 0) + ((j & 3) << 1);
                __half2 sB = *(const __half2*)&s_sb[m][Kl];
                __half2 cB = *(const __half2*)&s_cb[m][Kl];
                __half2 ct = *(const __half2*)&tcr[Kl];
                __half2 st = *(const __half2*)&tsr[Kl];
                A0.p[j] = __hfma2(sB, ct, __hmul2(cB, st));
                int K1 = Kl + 32;
                sB = *(const __half2*)&s_sb[m][K1];
                cB = *(const __half2*)&s_cb[m][K1];
                ct = *(const __half2*)&tcr[K1];
                st = *(const __half2*)&tsr[K1];
                A1.p[j] = __hfma2(sB, ct, __hmul2(cB, st));
            }

            float part[8];
#pragma unroll
            for (int i = 0; i < 8; ++i) part[i] = 0.f;

#pragma unroll
            for (int t = 0; t < 4; ++t) {
                v8f acc = {};
                acc = __builtin_amdgcn_wmma_f32_16x16x32_f16(false, A0.v, false, bf[t][0],
                                                             (short)0, acc, false, false);
                acc = __builtin_amdgcn_wmma_f32_16x16x32_f16(false, A1.v, false, bf[t][1],
                                                             (short)0, acc, false, false);
#pragma unroll
                for (int i = 0; i < 8; ++i) {
                    // C/D layout: M = i + 8*hi, N = ln + 16*t
                    float h1 = __sinf(OMEGA_K * (acc[i] + b1v[t]));
                    part[i] += h1 * s_u[rowb + i + hi * 8][t * 16 + ln];
                }
            }
            // reduce msg over N (16 lanes per half), scatter-add
#pragma unroll
            for (int i = 0; i < 8; ++i) {
                float v = part[i];
                v += __shfl_xor(v, 1);
                v += __shfl_xor(v, 2);
                v += __shfl_xor(v, 4);
                v += __shfl_xor(v, 8);
                if (ln == i) {
                    int M = i + hi * 8;
                    int d = s_dst[rowb + M];
                    if (d >= 0) atomicAdd(outacc + (size_t)d * H + c, v);
                }
            }
        }
    }
}

// ---------------------------------------------------------------- activation: sin(0.01*(acc+xbsum+bias))
__global__ void k_act(const float* __restrict__ acc, const float* __restrict__ xbsum,
                      const float* __restrict__ bias, float* __restrict__ hact, int N) {
    int id = blockIdx.x * blockDim.x + threadIdx.x;
    if (id >= N * H) return;
    int n = id >> 6, c = id & 63;
    hact[id] = __sinf(OMEGA_ENC * (acc[id] + xbsum[n] + bias[c]));
}

// ---------------------------------------------------------------- pool max (ordered-uint atomicMax)
__global__ void k_poolscatter(const int* __restrict__ srcb, const int* __restrict__ dstb,
                              const float* __restrict__ hact, unsigned* __restrict__ tmp, int Etot) {
    int id = blockIdx.x * blockDim.x + threadIdx.x;
    if (id >= Etot * H) return;
    int e = id >> 6, c = id & 63;
    float v = hact[(size_t)srcb[e] * H + c];
    unsigned b = __float_as_uint(v);
    unsigned enc = (b & 0x80000000u) ? ~b : (b | 0x80000000u);
    atomicMax(tmp + (size_t)dstb[e] * H + c, enc);
}

__global__ void k_gatherpool(const unsigned* __restrict__ tmp, const int* __restrict__ keep,
                             float* __restrict__ hp, int N1) {
    int id = blockIdx.x * blockDim.x + threadIdx.x;
    if (id >= N1 * H) return;
    int n = id >> 6, c = id & 63;
    unsigned enc = tmp[(size_t)keep[n] * H + c];
    unsigned b = (enc & 0x80000000u) ? (enc & 0x7FFFFFFFu) : ~enc;
    hp[id] = __uint_as_float(b);
}

// ---------------------------------------------------------------- final: sin(0.01*(hb@lin1_w+lin1_b))
__global__ void k_final(const float* __restrict__ hb, const float* __restrict__ w,
                        const float* __restrict__ b, float* __restrict__ out, int N1) {
    int id = blockIdx.x * blockDim.x + threadIdx.x;
    if (id >= N1 * LAT) return;
    int n = id >> 5, l = id & 31;
    float s = b[l];
#pragma unroll 8
    for (int h2 = 0; h2 < H; ++h2) s += hb[n * H + h2] * w[h2 * LAT + l];
    out[id] = __sinf(OMEGA_ENC * s);
}

// ================================================================ host launcher
extern "C" void kernel_launch(void* const* d_in, const int* in_sizes, int n_in,
                              void* d_out, int out_size, void* d_ws, size_t ws_size,
                              hipStream_t stream) {
    const float* x      = (const float*)d_in[0];
    const float* pos    = (const float*)d_in[1];
    const int*   ei0    = (const int*)  d_in[2];
    const int*   ei1    = (const int*)  d_in[3];
    const float* ppos1  = (const float*)d_in[4];
    const int*   keep   = (const int*)  d_in[5];
    const float* lin0_w = (const float*)d_in[6];
    const float* lin0_b = (const float*)d_in[7];
    const float* lin1_w = (const float*)d_in[8];
    const float* lin1_b = (const float*)d_in[9];
    const float* c0_w0 = (const float*)d_in[10]; const float* c0_b0 = (const float*)d_in[11];
    const float* c0_w1 = (const float*)d_in[12]; const float* c0_b1 = (const float*)d_in[13];
    const float* c0_w2 = (const float*)d_in[14]; const float* c0_b2 = (const float*)d_in[15];
    const float* c0_bias = (const float*)d_in[16];
    const float* c1_w0 = (const float*)d_in[17]; const float* c1_b0 = (const float*)d_in[18];
    const float* c1_w1 = (const float*)d_in[19]; const float* c1_b1 = (const float*)d_in[20];
    const float* c1_w2 = (const float*)d_in[21]; const float* c1_b2 = (const float*)d_in[22];
    const float* c1_bias = (const float*)d_in[23];
    float* out = (float*)d_out;

    const int N0 = in_sizes[1] / 3;           // 4096
    const int E0 = in_sizes[2] / 2;           // 16384
    const int E1 = in_sizes[3] / 2;           // 8192
    const int N1 = in_sizes[4] / 3;           // 1024
    const int Etot0 = E0 + N0;                // self loops appended
    const int Etot1 = E1 + N1;

    // workspace carve-up
    char* ws = (char*)d_ws;
    size_t off = 0;
    auto alloc = [&](size_t bytes) { char* p = ws + off; off += (bytes + 255) & ~(size_t)255; return p; };
    float*    henc    = (float*)alloc((size_t)N0 * H * 4);
    __half*   tabc0   = (__half*)alloc(H * H * 2);
    __half*   tabs0   = (__half*)alloc(H * H * 2);
    __half*   tabc1   = (__half*)alloc(H * H * 2);
    __half*   tabs1   = (__half*)alloc(H * H * 2);
    __half*   sb0     = (__half*)alloc((size_t)Etot0 * H * 2);
    __half*   cb0     = (__half*)alloc((size_t)Etot0 * H * 2);
    float*    u0      = (float*)alloc((size_t)Etot0 * H * 4);
    int*      dst0    = (int*)  alloc((size_t)Etot0 * 4);
    int*      src0    = (int*)  alloc((size_t)Etot0 * 4);
    float*    xbsum0  = (float*)alloc((size_t)N0 * 4);
    float*    acc0    = (float*)alloc((size_t)N0 * H * 4);
    float*    hact0   = (float*)alloc((size_t)N0 * H * 4);
    unsigned* pooltmp = (unsigned*)alloc((size_t)N0 * H * 4);
    float*    hp      = (float*)alloc((size_t)N1 * H * 4);
    __half*   sb1     = (__half*)alloc((size_t)Etot1 * H * 2);
    __half*   cb1     = (__half*)alloc((size_t)Etot1 * H * 2);
    float*    u1      = (float*)alloc((size_t)Etot1 * H * 4);
    int*      dst1    = (int*)  alloc((size_t)Etot1 * 4);
    int*      src1    = (int*)  alloc((size_t)Etot1 * 4);
    float*    xbsum1  = (float*)alloc((size_t)N1 * 4);
    float*    acc1    = (float*)alloc((size_t)N1 * H * 4);
    float*    hb1     = (float*)alloc((size_t)N1 * H * 4);
    (void)ws_size; (void)n_in; (void)out_size;

    auto gz = [](int n) { return (n + 255) / 256; };

    // zero accumulators (harness does not re-zero between replays)
    k_zero<<<gz(N0), 256, 0, stream>>>(xbsum0, N0);
    k_zero<<<gz(N0 * H), 256, 0, stream>>>(acc0, N0 * H);
    k_zero<<<gz(N0 * H), 256, 0, stream>>>((float*)pooltmp, N0 * H);
    k_zero<<<gz(N1), 256, 0, stream>>>(xbsum1, N1);
    k_zero<<<gz(N1 * H), 256, 0, stream>>>(acc1, N1 * H);

    k_encode<<<gz(N0 * H), 256, 0, stream>>>(x, lin0_w, lin0_b, henc, N0);
    k_tabs<<<gz(H * H), 256, 0, stream>>>(c0_w0, tabc0, tabs0);
    k_tabs<<<gz(H * H), 256, 0, stream>>>(c1_w0, tabc1, tabs1);

    // ---- level 0 conv
    k_edge_prep<<<Etot0, 64, 0, stream>>>(ei0, E0, Etot0, pos, henc,
                                          c0_w0, c0_b0, c0_w2, c0_b2,
                                          sb0, cb0, u0, dst0, src0, xbsum0);
    k_conv_wmma<<<(Etot0 + 31) / 32, 256, 0, stream>>>(sb0, cb0, u0, dst0, tabc0, tabs0,
                                                       c0_w1, c0_b1, acc0, Etot0);
    k_act<<<gz(N0 * H), 256, 0, stream>>>(acc0, xbsum0, c0_bias, hact0, N0);

    // ---- pooling
    k_poolscatter<<<gz(Etot0 * H), 256, 0, stream>>>(src0, dst0, hact0, pooltmp, Etot0);
    k_gatherpool<<<gz(N1 * H), 256, 0, stream>>>(pooltmp, keep, hp, N1);

    // ---- level 1 conv
    k_edge_prep<<<Etot1, 64, 0, stream>>>(ei1, E1, Etot1, ppos1, hp,
                                          c1_w0, c1_b0, c1_w2, c1_b2,
                                          sb1, cb1, u1, dst1, src1, xbsum1);
    k_conv_wmma<<<(Etot1 + 31) / 32, 256, 0, stream>>>(sb1, cb1, u1, dst1, tabc1, tabs1,
                                                       c1_w1, c1_b1, acc1, Etot1);
    k_act<<<gz(N1 * H), 256, 0, stream>>>(acc1, xbsum1, c1_bias, hb1, N1);

    // ---- decoder
    k_final<<<gz(N1 * LAT), 256, 0, stream>>>(hb1, lin1_w, lin1_b, out, N1);
}
